// SSFusionStateS2P_62620623176384
// MI455X (gfx1250) — matrix operand order repaired and checked by reading, not compiled, not run on hardware
//
#include <hip/hip_runtime.h>
#include <hip/hip_bf16.h>
#include <cstdint>

#define DI __device__ __forceinline__

typedef __attribute__((ext_vector_type(16))) _Float16 v16h;
typedef __attribute__((ext_vector_type(8)))  float    v8f;

namespace {

constexpr int kB  = 16;
constexpr int kN  = 4096;
constexpr int kC  = 512;
constexpr int kH  = 16;
constexpr int kHd = 32;
constexpr int kM  = kB * kN;   // 65536 rows

// ---- WMMA fragment loaders (CDNA5 ISA 7.12.2, 16-bit operands) ----
// A 16x32 row-major tile s[m*ld + k]:
//   lane (hi, m): K runs [hi*8, hi*8+8) and [16+hi*8, 16+hi*8+8)
//   -> two contiguous 16B chunks per lane (2x ds_load_b128)
DI v16h frag_a_rowmajor(const _Float16* s, int ld, int mOff) {
  int m  = mOff + (threadIdx.x & 15);
  int hi = (threadIdx.x >> 4) & 1;
  const _Float16* p0 = s + m * ld + hi * 8;
  const _Float16* p1 = p0 + 16;
  v16h a;
#pragma unroll
  for (int h = 0; h < 8; ++h) a[h] = p0[h];
#pragma unroll
  for (int h = 0; h < 8; ++h) a[8 + h] = p1[h];
  return a;
}
// B 32x16: lane (hi, n) needs K runs [hi*8, hi*8+8) and [16+hi*8, 16+hi*8+8) of column n.
// Tile staged SWIZZLED as s[n*ld + k] so the values are two contiguous
// 16B runs per lane (2x ds_load_b128).
DI v16h frag_b_swz(const _Float16* s, int ld, int nOff) {
  int n  = nOff + (threadIdx.x & 15);
  int hi = (threadIdx.x >> 4) & 1;
  const _Float16* p = s + n * ld + hi * 8;
  v16h b;
#pragma unroll
  for (int h = 0; h < 8; ++h) b[h] = p[h];
#pragma unroll
  for (int h = 0; h < 8; ++h) b[8 + h] = p[16 + h];
  return b;
}

DI v8f wmma_f16(v16h a, v16h b, v8f c) {
  return __builtin_amdgcn_wmma_f32_16x16x32_f16(
      false, a, false, b, (short)0, c, false, false);
}

// ============================================================================
// K1: qkv = x @ Wqkv + bqkv, fused epilogue:
//     q = relu(q)            -> q_h  [(b*H+h), n, d]  f16
//     k = relu(k + pos_enc)  -> k_h  [(b*H+h), n, d]  f16
//     v                      -> v_img[(b*H+h), d, n]  f16 (image-major for conv)
// 256 thr (8 waves), tile 128x128, K-step 32, double-buffered LDS.
// ============================================================================
__global__ __launch_bounds__(256) void qkv_gemm_kernel(
    const float* __restrict__ x, const float* __restrict__ Wqkv,
    const float* __restrict__ bqkv, const float* __restrict__ pos_enc,
    _Float16* __restrict__ q_h, _Float16* __restrict__ k_h,
    _Float16* __restrict__ v_img) {
  __shared__ __align__(16) _Float16 As[2][128 * 40];    // [m][k]
  __shared__ __align__(16) _Float16 Bsw[2][128 * 40];   // [n][k] swizzled

  const int t       = threadIdx.x;
  const int wave    = t >> 5;
  const int waveM   = wave >> 1;      // 0..3 -> +32 rows
  const int waveN   = wave & 1;       // 0..1 -> +64 cols
  const int rowBase = blockIdx.x * 128;
  const int colBase = blockIdx.y * 128;

  v8f zero = {};
  v8f acc[2][4];
#pragma unroll
  for (int mt = 0; mt < 2; ++mt)
#pragma unroll
    for (int nt = 0; nt < 4; ++nt) acc[mt][nt] = zero;

  const int ar = t >> 1, ah = t & 1;    // A staging: row, 16-col half
  const int bk = t >> 3, bsg = t & 7;   // B staging: k-row, 16-col segment

  float4 aReg[4], bReg[4];
  auto loadTiles = [&](int k0) {
    const float* aSrc = x + (size_t)(rowBase + ar) * kC + k0 + ah * 16;
#pragma unroll
    for (int i = 0; i < 4; ++i) aReg[i] = ((const float4*)aSrc)[i];
    const float* bSrc = Wqkv + (size_t)(k0 + bk) * (3 * kC) + colBase + bsg * 16;
#pragma unroll
    for (int i = 0; i < 4; ++i) bReg[i] = ((const float4*)bSrc)[i];
  };
  auto storeTiles = [&](int buf) {
#pragma unroll
    for (int i = 0; i < 4; ++i) {
      _Float16* pa = &As[buf][ar * 40 + ah * 16 + i * 4];
      pa[0] = (_Float16)aReg[i].x; pa[1] = (_Float16)aReg[i].y;
      pa[2] = (_Float16)aReg[i].z; pa[3] = (_Float16)aReg[i].w;
    }
    const float* bf = (const float*)bReg;
#pragma unroll
    for (int i = 0; i < 16; ++i)
      Bsw[buf][(bsg * 16 + i) * 40 + bk] = (_Float16)bf[i];
  };

  loadTiles(0);
  storeTiles(0);
  __syncthreads();

  for (int k0 = 0; k0 < kC; k0 += 32) {
    const int  cur     = (k0 >> 5) & 1;
    const bool hasNext = (k0 + 32) < kC;
    if (hasNext) loadTiles(k0 + 32);
    if (k0 + 64 < kC) {
      __builtin_prefetch(x + (size_t)(rowBase + ar) * kC + k0 + 64 + ah * 16, 0, 1);
      __builtin_prefetch(Wqkv + (size_t)(k0 + 64 + bk) * (3 * kC) + colBase + bsg * 16, 0, 1);
    }

    v16h aF[2], bF[4];
#pragma unroll
    for (int mt = 0; mt < 2; ++mt)
      aF[mt] = frag_a_rowmajor(As[cur], 40, waveM * 32 + mt * 16);
#pragma unroll
    for (int nt = 0; nt < 4; ++nt)
      bF[nt] = frag_b_swz(Bsw[cur], 40, waveN * 64 + nt * 16);
#pragma unroll
    for (int mt = 0; mt < 2; ++mt)
#pragma unroll
      for (int nt = 0; nt < 4; ++nt)
        acc[mt][nt] = wmma_f16(aF[mt], bF[nt], acc[mt][nt]);

    if (hasNext) storeTiles(cur ^ 1);
    __syncthreads();
  }

  // Epilogue: D element (m = hi*8+r, n = lane&15)
  const int lnN = t & 15, lnHi = (t >> 4) & 1;
#pragma unroll
  for (int mt = 0; mt < 2; ++mt) {
#pragma unroll
    for (int nt = 0; nt < 4; ++nt) {
#pragma unroll
      for (int r = 0; r < 8; ++r) {
        int m   = lnHi * 8 + r;
        int row = rowBase + waveM * 32 + mt * 16 + m;
        int col = colBase + waveN * 64 + nt * 16 + lnN;
        float val = acc[mt][nt][r] + bqkv[col];
        int b = row >> 12, nseq = row & (kN - 1);
        int sec = col >> 9, cc = col & (kC - 1);
        int hh = cc >> 5, d = cc & 31;
        int bh = b * kH + hh;
        if (sec == 0) {
          val = fmaxf(val, 0.f);
          q_h[((size_t)bh * kN + nseq) * kHd + d] = (_Float16)val;
        } else if (sec == 1) {
          val += pos_enc[(size_t)nseq * kC + cc];
          val = fmaxf(val, 0.f);
          k_h[((size_t)bh * kN + nseq) * kHd + d] = (_Float16)val;
        } else {
          v_img[((size_t)bh * kHd + d) * kN + nseq] = (_Float16)val;
        }
      }
    }
  }
}

// ============================================================================
// K2: ksum[bh,d] = sum_n k_h[bh,n,d]  — 128-bit chunked loads.
// Thread (rr = t>>2, part = t&3) accumulates 8 dims over rows rr, rr+64, ...
// ============================================================================
__global__ __launch_bounds__(256) void ksum_kernel(
    const _Float16* __restrict__ k_h, float* __restrict__ ksum) {
  __shared__ float red[256 * 8];
  const int bh = blockIdx.x;
  const int t  = threadIdx.x;
  const int rr = t >> 2, part = t & 3;
  float acc[8];
#pragma unroll
  for (int j = 0; j < 8; ++j) acc[j] = 0.f;
  for (int r = rr; r < kN; r += 64) {
    const _Float16* src = k_h + ((size_t)bh * kN + r) * kHd + part * 8;
    _Float16 tmp[8];
    *(uint4*)tmp = *(const uint4*)src;
#pragma unroll
    for (int j = 0; j < 8; ++j) acc[j] += (float)tmp[j];
  }
#pragma unroll
  for (int j = 0; j < 8; ++j) red[t * 8 + j] = acc[j];
  __syncthreads();
  if (t < 32) {
    int p2 = t >> 3, j = t & 7;
    float s = 0.f;
    for (int r2 = 0; r2 < 64; ++r2) s += red[(r2 * 4 + p2) * 8 + j];
    ksum[bh * kHd + p2 * 8 + j] = s;
  }
}

// ============================================================================
// K3: kv[bh] = k^T v  (32x32, K=4096) via WMMA, double-buffered.
// k tile staged TRANSPOSED -> KtT[c][j] so A frags are contiguous;
// v_img is already [d][j] so B frags are contiguous.
// 256 blocks x 128 thr (4 waves), one 16x16 tile per wave.
// ============================================================================
__global__ __launch_bounds__(128) void kv_gemm_kernel(
    const _Float16* __restrict__ k_h, const _Float16* __restrict__ v_img,
    float* __restrict__ kv) {
  __shared__ __align__(16) _Float16 KtT[2][32 * 40];   // [c][j]
  __shared__ __align__(16) _Float16 VtT[2][32 * 40];   // [d][j]
  const int bh = blockIdx.x;
  const int t  = threadIdx.x;
  const int wave = t >> 5;
  const int c0 = (wave >> 1) * 16, d0 = (wave & 1) * 16;

  const int kj = t >> 2, kcp = (t & 3) * 8;  // k chunk: row j, 8-dim part
  const int vd = t >> 2, vp  = t & 3;        // v chunk: dim d, 8-j part
  const int ln = t & 15, hi  = (t >> 4) & 1;

  uint4 kChunk, vChunk;
  auto loadChunks = [&](int j0) {
    kChunk = *(const uint4*)(k_h + ((size_t)bh * kN + j0 + kj) * kHd + kcp);
    vChunk = *(const uint4*)(v_img + ((size_t)bh * kHd + vd) * kN + j0 + vp * 8);
  };
  auto storeChunks = [&](int buf) {
    _Float16 tmp[8];
    *(uint4*)tmp = kChunk;
#pragma unroll
    for (int i = 0; i < 8; ++i) KtT[buf][(kcp + i) * 40 + kj] = tmp[i];
    *(uint4*)(&VtT[buf][vd * 40 + vp * 8]) = vChunk;
  };

  v8f acc = {};
  loadChunks(0);
  storeChunks(0);
  __syncthreads();
  for (int j0 = 0; j0 < kN; j0 += 32) {
    const int  cur     = (j0 >> 5) & 1;
    const bool hasNext = (j0 + 32) < kN;
    if (hasNext) loadChunks(j0 + 32);

    v16h a = frag_a_rowmajor(KtT[cur], 40, c0);
    v16h b = frag_b_swz(VtT[cur], 40, d0);
    acc = wmma_f16(a, b, acc);

    if (hasNext) storeChunks(cur ^ 1);
    __syncthreads();
  }
#pragma unroll
  for (int r = 0; r < 8; ++r) {
    int m = hi * 8 + r;
    kv[(size_t)bh * kHd * kHd + (c0 + m) * kHd + d0 + ln] = acc[r];
  }
}

// ============================================================================
// K4: depthwise 5x5 conv (pad 2) on v_img; f32 LDS halo tile (no per-tap cvt).
// One block per channel (bh,d); fm_h written row-major [(bh), n, d] f16.
// ============================================================================
__global__ __launch_bounds__(256) void dwconv_kernel(
    const _Float16* __restrict__ v_img, const float* __restrict__ dwc_w,
    const float* __restrict__ dwc_b, _Float16* __restrict__ fm_h) {
  __shared__ float tile[68 * 68];
  const int ch = blockIdx.x;          // bh*32 + d
  const int bh = ch >> 5, d = ch & 31;
  const int t  = threadIdx.x;

  for (int i = t; i < 68 * 68; i += 256) tile[i] = 0.f;
  __syncthreads();
  {
    const int y = t >> 2, part = t & 3;
    const _Float16* src = v_img + (size_t)ch * kN + y * 64 + part * 16;
    _Float16 tmp[16];
    *(uint4*)(tmp)     = ((const uint4*)src)[0];
    *(uint4*)(tmp + 8) = ((const uint4*)src)[1];
#pragma unroll
    for (int i = 0; i < 16; ++i)
      tile[(y + 2) * 68 + 2 + part * 16 + i] = (float)tmp[i];
  }
  __syncthreads();

  float w[25];
#pragma unroll
  for (int i = 0; i < 25; ++i) w[i] = dwc_w[d * 25 + i];
  const float bias = dwc_b[d];

#pragma unroll 4
  for (int i = 0; i < 16; ++i) {
    int p = t * 16 + i;
    int y = p >> 6, xx = p & 63;
    float a = bias;
#pragma unroll
    for (int dy = 0; dy < 5; ++dy)
#pragma unroll
      for (int dx = 0; dx < 5; ++dx)
        a += w[dy * 5 + dx] * tile[(y + dy) * 68 + xx + dx];
    fm_h[((size_t)bh * kN + p) * kHd + d] = (_Float16)a;
  }
}

// ============================================================================
// K5: out = (q @ kv) * z + fm, z_i = 1/(q_i . ksum + 1e-6); writes merged-head
// activation act[(b*N+n), h*hd+d] f16. kv staged swizzled [d][c] for B frags.
// Grid (32 row-blocks, 256 heads), 256 thr (8 waves x 16 rows), one K-step.
// ============================================================================
__global__ __launch_bounds__(256) void attn_out_kernel(
    const _Float16* __restrict__ q_h, const float* __restrict__ kv,
    const float* __restrict__ ksum, const _Float16* __restrict__ fm_h,
    _Float16* __restrict__ act) {
  __shared__ __align__(16) _Float16 kvs[32 * 40];    // [d][c] swizzled
  __shared__ float    ksl[32];
  __shared__ __align__(16) _Float16 qs[8][16 * 40];  // per-wave q rows
  __shared__ float    zs[8][16];

  const int bh = blockIdx.y;
  const int t  = threadIdx.x;
  const int wave = t >> 5;
  const int ln = t & 15, hi = (t >> 4) & 1;
  const int i0 = blockIdx.x * 128 + wave * 16;

  {
    const float* src = kv + (size_t)bh * kHd * kHd;
#pragma unroll
    for (int j = 0; j < 4; ++j) {
      int idx = t * 4 + j;                 // idx = c*32 + d
      kvs[(idx & 31) * 40 + (idx >> 5)] = (_Float16)src[idx];
    }
    if (t < 32) ksl[t] = ksum[bh * kHd + t];
  }
  {
    int lane = t & 31;
    int row = lane >> 1, half = lane & 1;
    const _Float16* src = q_h + ((size_t)bh * kN + i0 + row) * kHd + half * 16;
    _Float16 tmp[16];
    *(uint4*)(tmp)     = ((const uint4*)src)[0];
    *(uint4*)(tmp + 8) = ((const uint4*)src)[1];
#pragma unroll
    for (int i = 0; i < 16; ++i) qs[wave][row * 40 + half * 16 + i] = tmp[i];
  }
  __syncthreads();

  if ((t & 31) < 16) {
    int m = t & 15;
    float s = 0.f;
#pragma unroll
    for (int c = 0; c < kHd; ++c) s += (float)qs[wave][m * 40 + c] * ksl[c];
    zs[wave][m] = 1.f / (s + 1e-6f);
  }
  __syncthreads();

  v16h a  = frag_a_rowmajor(qs[wave], 40, 0);
  v16h b0 = frag_b_swz(kvs, 40, 0);
  v16h b1 = frag_b_swz(kvs, 40, 16);
  v8f acc0 = {}, acc1 = {};
  acc0 = wmma_f16(a, b0, acc0);
  acc1 = wmma_f16(a, b1, acc1);

  const int b = bh >> 4, hh = bh & 15;
#pragma unroll
  for (int r = 0; r < 8; ++r) {
    int m = hi * 8 + r;
    int i = i0 + m;
    float z = zs[wave][m];
    float f0 = (float)fm_h[((size_t)bh * kN + i) * kHd + ln];
    float f1 = (float)fm_h[((size_t)bh * kN + i) * kHd + 16 + ln];
    size_t base = ((size_t)(b * kN + i)) * kC + hh * kHd;
    act[base + ln]      = (_Float16)(acc0[r] * z + f0);
    act[base + 16 + ln] = (_Float16)(acc1[r] * z + f1);
  }
}

// ============================================================================
// K6: out = act @ Wproj + bproj  (65536x512 @ 512x512, fp32 out)
// Same double-buffered, B-swizzled tiling as K1.
// ============================================================================
__global__ __launch_bounds__(256) void proj_gemm_kernel(
    const _Float16* __restrict__ act, const float* __restrict__ Wproj,
    const float* __restrict__ bproj, float* __restrict__ out) {
  __shared__ __align__(16) _Float16 As[2][128 * 40];
  __shared__ __align__(16) _Float16 Bsw[2][128 * 40];

  const int t = threadIdx.x;
  const int wave = t >> 5;
  const int waveM = wave >> 1, waveN = wave & 1;
  const int rowBase = blockIdx.x * 128;
  const int colBase = blockIdx.y * 128;

  v8f zero = {};
  v8f acc[2][4];
#pragma unroll
  for (int mt = 0; mt < 2; ++mt)
#pragma unroll
    for (int nt = 0; nt < 4; ++nt) acc[mt][nt] = zero;

  const int ar = t >> 1, ah = t & 1;
  const int bk = t >> 3, bsg = t & 7;

  uint4 a0, a1;
  float4 bReg[4];
  auto loadTiles = [&](int k0) {
    const _Float16* aSrc = act + (size_t)(rowBase + ar) * kC + k0 + ah * 16;
    a0 = ((const uint4*)aSrc)[0];
    a1 = ((const uint4*)aSrc)[1];
    const float* bSrc = Wproj + (size_t)(k0 + bk) * kC + colBase + bsg * 16;
#pragma unroll
    for (int i = 0; i < 4; ++i) bReg[i] = ((const float4*)bSrc)[i];
  };
  auto storeTiles = [&](int buf) {
    *(uint4*)(&As[buf][ar * 40 + ah * 16])     = a0;
    *(uint4*)(&As[buf][ar * 40 + ah * 16 + 8]) = a1;
    const float* bf = (const float*)bReg;
#pragma unroll
    for (int i = 0; i < 16; ++i)
      Bsw[buf][(bsg * 16 + i) * 40 + bk] = (_Float16)bf[i];
  };

  loadTiles(0);
  storeTiles(0);
  __syncthreads();

  for (int k0 = 0; k0 < kC; k0 += 32) {
    const int  cur     = (k0 >> 5) & 1;
    const bool hasNext = (k0 + 32) < kC;
    if (hasNext) loadTiles(k0 + 32);
    if (k0 + 64 < kC) {
      __builtin_prefetch(act + (size_t)(rowBase + ar) * kC + k0 + 64 + ah * 16, 0, 1);
      __builtin_prefetch(Wproj + (size_t)(k0 + 64 + bk) * kC + colBase + bsg * 16, 0, 1);
    }

    v16h aF[2], bF[4];
#pragma unroll
    for (int mt = 0; mt < 2; ++mt)
      aF[mt] = frag_a_rowmajor(As[cur], 40, waveM * 32 + mt * 16);
#pragma unroll
    for (int nt = 0; nt < 4; ++nt)
      bF[nt] = frag_b_swz(Bsw[cur], 40, waveN * 64 + nt * 16);
#pragma unroll
    for (int mt = 0; mt < 2; ++mt)
#pragma unroll
      for (int nt = 0; nt < 4; ++nt)
        acc[mt][nt] = wmma_f16(aF[mt], bF[nt], acc[mt][nt]);

    if (hasNext) storeTiles(cur ^ 1);
    __syncthreads();
  }

  const int lnN = t & 15, lnHi = (t >> 4) & 1;
#pragma unroll
  for (int mt = 0; mt < 2; ++mt) {
#pragma unroll
    for (int nt = 0; nt < 4; ++nt) {
#pragma unroll
      for (int r = 0; r < 8; ++r) {
        int row = rowBase + waveM * 32 + mt * 16 + lnHi * 8 + r;
        int col = colBase + waveN * 64 + nt * 16 + lnN;
        out[(size_t)row * kC + col] = acc[mt][nt][r] + bproj[col];
      }
    }
  }
}

}  // namespace

extern "C" void kernel_launch(void* const* d_in, const int* in_sizes, int n_in,
                              void* d_out, int out_size, void* d_ws, size_t ws_size,
                              hipStream_t stream) {
  (void)in_sizes; (void)n_in; (void)out_size; (void)ws_size;
  const float* x       = (const float*)d_in[0];
  const float* Wqkv    = (const float*)d_in[1];
  const float* bqkv    = (const float*)d_in[2];
  const float* pos_enc = (const float*)d_in[3];
  const float* dwc_w   = (const float*)d_in[4];
  const float* dwc_b   = (const float*)d_in[5];
  const float* Wproj   = (const float*)d_in[6];
  const float* bproj   = (const float*)d_in[7];
  float* out = (float*)d_out;

  char* p = (char*)d_ws;
  auto take = [&](size_t bytes) -> char* {
    char* r = p;
    p += (bytes + 255) & ~(size_t)255;
    return r;
  };
  const size_t nAct = (size_t)kB * kH * kN * kHd;  // 33.5M elems
  _Float16* q_h   = (_Float16*)take(nAct * sizeof(_Float16));
  _Float16* k_h   = (_Float16*)take(nAct * sizeof(_Float16));
  _Float16* v_img = (_Float16*)take(nAct * sizeof(_Float16));
  _Float16* fm_h  = (_Float16*)take(nAct * sizeof(_Float16));
  _Float16* act   = (_Float16*)take(nAct * sizeof(_Float16));
  float* ksum = (float*)take((size_t)kB * kH * kHd * sizeof(float));
  float* kv   = (float*)take((size_t)kB * kH * kHd * kHd * sizeof(float));

  qkv_gemm_kernel<<<dim3(kM / 128, (3 * kC) / 128), 256, 0, stream>>>(
      x, Wqkv, bqkv, pos_enc, q_h, k_h, v_img);
  ksum_kernel<<<kB * kH, 256, 0, stream>>>(k_h, ksum);
  kv_gemm_kernel<<<kB * kH, 128, 0, stream>>>(k_h, v_img, kv);
  dwconv_kernel<<<kB * kH * kHd, 256, 0, stream>>>(v_img, dwc_w, dwc_b, fm_h);
  attn_out_kernel<<<dim3(kN / 128, kB * kH), 256, 0, stream>>>(
      q_h, kv, ksum, fm_h, act);
  proj_gemm_kernel<<<dim3(kM / 128, kC / 128), 256, 0, stream>>>(
      act, Wproj, bproj, out);
}
